// LinearDecoder_81707457839462
// MI455X (gfx1250) — compile-verified, matrix operand order checked by reading, not run on hardware
//
#include <hip/hip_runtime.h>

// ---------------------------------------------------------------------------
// Edge-pair MLP decoder:  out = relu(concat(z[r],z[c]) @ W1^T + b1) @ W2^T + b2
// E=500000, D_Z=128, IN=256, HID=512, OUT=2.
//
// MI455X plan:
//  * bf16 WMMA (f32 acc). Each wave owns 32 edges (2 row-tiles) -> every
//    B operand feeds 2 WMMAs.
//  * W1 B-tiles (16x256 bf16 = 8KB) staged into LDS once per block via the
//    Tensor Data Mover (tensor_load_to_lds + s_wait_tensorcnt), double
//    buffered, shared by all 4 waves. Sync-copy fallback if builtin absent.
//  * Layer 2 fused: every 2 N-tiles one K=32 WMMA accumulation (B = W2).
//  * If workspace allows, z is pre-converted to bf16 so the A gather is pure
//    global_load_b128 in WMMA layout (no conversion VALU, half the bytes).
// ---------------------------------------------------------------------------

#define D_Z    128
#define IN_CH  256
#define HID    512
#define OUT_CH 2
#define NT     (HID / 16)      // 32 N-tiles of layer 1
#define BSTR   264             // B tile LDS row stride (bf16): 256 + 8 (TDM pad 4dw/128dw)
#define HSTR   40              // H chunk LDS row stride (bf16): 32 + 8
#define WAVES  4
#define TPW    2               // edge row-tiles per wave (M = 32 edges/wave)

typedef __attribute__((ext_vector_type(16))) __bf16 v16bf;
typedef __attribute__((ext_vector_type(8)))  float  v8f;
typedef __attribute__((ext_vector_type(4)))  unsigned int u32x4;
typedef __attribute__((ext_vector_type(4)))  int i32x4;
typedef __attribute__((ext_vector_type(8)))  int i32x8;

union V16 {
  v16bf v;
  unsigned int u[8];
  uint4 q[2];
};

struct SmemLayout {
  unsigned short B[2][16 * BSTR];          // double-buffered W1 tile, 2 x 8448 B
  unsigned short H[WAVES][TPW][16 * HSTR]; // per-wave h chunks, 4*2*1280 B
};

#if defined(__gfx1250__) && __has_builtin(__builtin_amdgcn_tensor_load_to_lds)
#define USE_TDM 1
#else
#define USE_TDM 0
#endif

__device__ __forceinline__ unsigned int pack2_bf16_rne(float a, float b) {
  unsigned int ua = __builtin_bit_cast(unsigned int, a);
  unsigned int ub = __builtin_bit_cast(unsigned int, b);
  ua = (ua + 0x7FFFu + ((ua >> 16) & 1u)) >> 16;
  ub = (ub + 0x7FFFu + ((ub >> 16) & 1u)) >> 16;
  return (ub << 16) | (ua & 0xFFFFu);
}

__device__ __forceinline__ unsigned short f32_to_bf16(float a) {
  unsigned int ua = __builtin_bit_cast(unsigned int, a);
  ua = (ua + 0x7FFFu + ((ua >> 16) & 1u)) >> 16;
  return (unsigned short)ua;
}

// One v_perm_b32: [bf16_trunc(a) | bf16_trunc(b)<<16]
__device__ __forceinline__ unsigned int pack2_bf16_trunc(float a, float b) {
  return __builtin_amdgcn_perm(__builtin_bit_cast(unsigned int, b),
                               __builtin_bit_cast(unsigned int, a),
                               0x07060302u);
}

__device__ __forceinline__ void wait_tensorcnt0() {
#if __has_builtin(__builtin_amdgcn_s_wait_tensorcnt)
  __builtin_amdgcn_s_wait_tensorcnt(0);
#else
  asm volatile("s_wait_tensorcnt 0x0" ::: "memory");
#endif
}

#if USE_TDM
// TDM load of W1 rows [n0, n0+16) x 256 bf16 into LDS (row stride 264).
// D# groups per CDNA5 ISA.
__device__ __forceinline__ void tdm_stage_b(const unsigned short* W1b, int n0,
                                            unsigned short* dstLds) {
  unsigned long long ga = (unsigned long long)(const void*)(W1b + (size_t)n0 * IN_CH);
  unsigned int lds_addr = (unsigned int)(unsigned long long)(const void*)dstLds;
  u32x4 g0;
  g0[0] = 1u;                                      // count=1 (valid user descriptor)
  g0[1] = lds_addr;                                // LDS byte address
  g0[2] = (unsigned int)ga;                        // global_addr[31:0]
  g0[3] = (unsigned int)(ga >> 32) | (2u << 30);   // global_addr[56:32] | type=2
  i32x8 g1;
  g1[0] = (int)0x07910000u;    // data_size=2B, pad_enable, pad_interval=128dw, pad_amount=4dw
  g1[1] = (int)(IN_CH << 16);  // tensor_dim0 = 256 (bits 63:48)
  g1[2] = (int)(HID << 16);    // tensor_dim1 = 512 (bits 95:80)
  g1[3] = (int)(IN_CH << 16);  // tile_dim0 = 256   (bits 127:112)
  g1[4] = 16;                  // tile_dim1 = 16    (bits 143:128)
  g1[5] = IN_CH;               // tensor_dim0_stride = 256 (bits 207:160)
  g1[6] = 0;
  g1[7] = 0;
  i32x4 gz = {0, 0, 0, 0};
#if __clang_major__ >= 23
  i32x8 gz8 = {0, 0, 0, 0, 0, 0, 0, 0};
  __builtin_amdgcn_tensor_load_to_lds(g0, g1, gz, gz, gz8, 0);
#else
  __builtin_amdgcn_tensor_load_to_lds(g0, g1, gz, gz, 0);
#endif
}
#else
__device__ __forceinline__ void sync_stage_b(const unsigned short* W1b, int n0,
                                             unsigned short* dst, int t) {
  #pragma unroll
  for (int k = 0; k < 4; ++k) {
    const int ch = t + k * 128;          // 512 chunks of 16B: 16 rows x 32 chunks
    const int row = ch >> 5, c16 = ch & 31;
    const uint4 v = *(const uint4*)(W1b + (size_t)(n0 + row) * IN_CH + c16 * 8);
    *(uint4*)(dst + row * BSTR + c16 * 8) = v;
  }
}
#endif

// Prologue: convert weights (and optionally z) to bf16 in workspace.
__global__ void cvt_weights_kernel(const float* __restrict__ W1,
                                   const float* __restrict__ W2,
                                   const float* __restrict__ z,
                                   unsigned short* __restrict__ W1b,
                                   unsigned short* __restrict__ W2b,
                                   unsigned short* __restrict__ zb,
                                   int n_z) {
  const int stride = gridDim.x * blockDim.x;
  const int i0 = blockIdx.x * blockDim.x + threadIdx.x;
  for (int t = i0; t < HID * IN_CH; t += stride) W1b[t] = f32_to_bf16(W1[t]);
  for (int t = i0; t < OUT_CH * HID; t += stride) W2b[t] = f32_to_bf16(W2[t]);
  if (zb) {
    for (int t = i0; t < n_z; t += stride) zb[t] = f32_to_bf16(z[t]);
  }
}

template <bool ZB>
__global__ void __launch_bounds__(WAVES * 32)
decoder_kernel(const float* __restrict__ z,
               const unsigned short* __restrict__ zb,
               const int* __restrict__ batch_r,
               const int* __restrict__ batch_c,
               const unsigned short* __restrict__ W1b,
               const float* __restrict__ b1,
               const unsigned short* __restrict__ W2b,
               const float* __restrict__ b2,
               float* __restrict__ out,
               int n_edges) {
  __shared__ SmemLayout smem;

  const int lane = threadIdx.x & 31;
  const int wave = threadIdx.x >> 5;
  const int m  = lane & 15;   // A row (edge) / B,C column
  const int hi = lane >> 4;   // K-subgroup select {0..7,16..23} vs {8..15,24..31}
  const int n_tiles = (n_edges + 15) >> 4;
  const int tile0 = blockIdx.x * (WAVES * TPW) + wave * TPW;

  int act[TPW];               // wave-uniform, forced scalar for clean s_cbranch + EXEC
  int ebase[TPW];
  V16 A[TPW][8];

  // ---- gather A tiles: 32 edges x 256 features, straight into WMMA A layout.
  #pragma unroll
  for (int t = 0; t < TPW; ++t) {
    act[t] = __builtin_amdgcn_readfirstlane((int)((tile0 + t) < n_tiles));
    ebase[t] = (tile0 + t) << 4;
    if (!act[t]) continue;
    int e = ebase[t] + m;
    if (e >= n_edges) e = n_edges - 1;
    const int ridx = batch_r[e];
    const int cidx = batch_c[e];
    #pragma unroll
    for (int ks = 0; ks < 8; ++ks) {
      #pragma unroll
      for (int g = 0; g < 2; ++g) {
        const int K = ks * 32 + g * 16 + hi * 8;   // 8 consecutive input features
        if (ZB) {
          const unsigned short* src = (K < D_Z) ? (zb + (size_t)ridx * D_Z + K)
                                                : (zb + (size_t)cidx * D_Z + (K - D_Z));
          A[t][ks].q[g] = *(const uint4*)src;
        } else {
          const float* src = (K < D_Z) ? (z + (size_t)ridx * D_Z + K)
                                       : (z + (size_t)cidx * D_Z + (K - D_Z));
          const float4 f0 = *(const float4*)(src);
          const float4 f1 = *(const float4*)(src + 4);
          A[t][ks].u[g * 4 + 0] = pack2_bf16_rne(f0.x, f0.y);
          A[t][ks].u[g * 4 + 1] = pack2_bf16_rne(f0.z, f0.w);
          A[t][ks].u[g * 4 + 2] = pack2_bf16_rne(f1.x, f1.y);
          A[t][ks].u[g * 4 + 3] = pack2_bf16_rne(f1.z, f1.w);
        }
      }
    }
  }

  v8f acc2[TPW];
  #pragma unroll
  for (int t = 0; t < TPW; ++t) acc2[t] = (v8f){};

  // ---- stage first B tile ----
#if USE_TDM
  if (wave == 0) tdm_stage_b(W1b, 0, &smem.B[0][0]);
#else
  sync_stage_b(W1b, 0, &smem.B[0][0], threadIdx.x);
#endif

  for (int nt = 0; nt < NT; ++nt) {
#if USE_TDM
    if (wave == 0) wait_tensorcnt0();     // stage nt complete (TENSORcnt is per-wave)
#endif
    __syncthreads();                      // B[nt&1] visible; prior reads of other buffer done

    if (nt + 1 < NT) {
#if USE_TDM
      if (wave == 0) tdm_stage_b(W1b, (nt + 1) * 16, &smem.B[(nt + 1) & 1][0]);
#else
      sync_stage_b(W1b, (nt + 1) * 16, &smem.B[(nt + 1) & 1][0], threadIdx.x);
#endif
    }

    // ---- layer 1: read B operands from LDS once, feed both edge tiles ----
    const unsigned short* Bt = &smem.B[nt & 1][0];
    V16 B[8];
    #pragma unroll
    for (int ks = 0; ks < 8; ++ks) {
      const unsigned short* brow = Bt + m * BSTR + ks * 32 + hi * 8;
      B[ks].q[0] = *(const uint4*)(brow);
      B[ks].q[1] = *(const uint4*)(brow + 16);
    }
    const float bias = b1[nt * 16 + m];

    #pragma unroll
    for (int t = 0; t < TPW; ++t) {
      if (!act[t]) continue;
      v8f acc = {};
      #pragma unroll
      for (int ks = 0; ks < 8; ++ks) {
        acc = __builtin_amdgcn_wmma_f32_16x16x32_bf16(false, A[t][ks].v, false, B[ks].v,
                                                      (short)0, acc, false, false);
      }
      // C layout: VGPR v, lane -> row = v + 8*hi, col = m. Bias+ReLU -> bf16 h chunk.
      unsigned short* hbase = &smem.H[wave][t][(nt & 1) * 16 + m + hi * 8 * HSTR];
      #pragma unroll
      for (int v = 0; v < 8; v += 2) {
        const float h0 = fmaxf(acc[v] + bias, 0.0f);
        const float h1 = fmaxf(acc[v + 1] + bias, 0.0f);
        const unsigned int p = pack2_bf16_trunc(h0, h1);   // one v_perm_b32
        hbase[v * HSTR]       = (unsigned short)p;         // ds_store_b16
        hbase[(v + 1) * HSTR] = (unsigned short)(p >> 16); // ds_store_b16_d16_hi
      }
    }

    // ---- layer 2: every 2 N-tiles, one K=32 WMMA accumulation per edge tile ----
    if (nt & 1) {
      V16 B2;
      if (m < OUT_CH) {
        const unsigned short* w2row = W2b + (size_t)m * HID + (nt >> 1) * 32 + hi * 8;
        B2.q[0] = *(const uint4*)(w2row);
        B2.q[1] = *(const uint4*)(w2row + 16);
      } else {
        B2.q[0] = make_uint4(0u, 0u, 0u, 0u);
        B2.q[1] = make_uint4(0u, 0u, 0u, 0u);
      }
      #pragma unroll
      for (int t = 0; t < TPW; ++t) {
        if (!act[t]) continue;
        V16 A2;
        const unsigned short* hp = &smem.H[wave][t][m * HSTR + hi * 8];
        A2.q[0] = *(const uint4*)(hp);
        A2.q[1] = *(const uint4*)(hp + 16);
        acc2[t] = __builtin_amdgcn_wmma_f32_16x16x32_bf16(false, A2.v, false, B2.v,
                                                          (short)0, acc2[t], false, false);
      }
    }
  }

  // ---- write out: only columns 0..1 are real ----
  if (m < OUT_CH) {
    const float bias2 = b2[m];
    #pragma unroll
    for (int t = 0; t < TPW; ++t) {
      if (!act[t]) continue;
      #pragma unroll
      for (int v = 0; v < 8; ++v) {
        const int eo = ebase[t] + v + hi * 8;
        if (eo < n_edges) out[(size_t)eo * OUT_CH + m] = acc2[t][v] + bias2;
      }
    }
  }
}

extern "C" void kernel_launch(void* const* d_in, const int* in_sizes, int n_in,
                              void* d_out, int out_size, void* d_ws, size_t ws_size,
                              hipStream_t stream) {
  const float* z  = (const float*)d_in[0];
  const int* br   = (const int*)d_in[1];
  const int* bc   = (const int*)d_in[2];
  const float* W1 = (const float*)d_in[3];
  const float* b1 = (const float*)d_in[4];
  const float* W2 = (const float*)d_in[5];
  const float* b2 = (const float*)d_in[6];
  float* out      = (float*)d_out;
  const int n_edges = in_sizes[1];
  const int n_z     = in_sizes[0];                 // 100000 * 128

  unsigned short* W1b = (unsigned short*)d_ws;                 // 512*256 bf16
  unsigned short* W2b = W1b + (size_t)HID * IN_CH;             // 2*512 bf16
  unsigned short* zb  = W2b + (size_t)OUT_CH * HID;            // optional z in bf16

  const size_t need = ((size_t)HID * IN_CH + (size_t)OUT_CH * HID + (size_t)n_z)
                      * sizeof(unsigned short);
  const bool use_zb = (ws_size >= need);           // deterministic per-deployment choice

  cvt_weights_kernel<<<use_zb ? 1024 : 64, 256, 0, stream>>>(
      W1, W2, z, W1b, W2b, use_zb ? zb : (unsigned short*)nullptr, n_z);

  const int n_tiles = (n_edges + 15) >> 4;
  const int tiles_per_block = WAVES * TPW;         // 8 tiles = 128 edges
  const int blocks = (n_tiles + tiles_per_block - 1) / tiles_per_block;
  if (use_zb) {
    decoder_kernel<true><<<blocks, WAVES * 32, 0, stream>>>(
        z, zb, br, bc, W1b, b1, W2b, b2, out, n_edges);
  } else {
    decoder_kernel<false><<<blocks, WAVES * 32, 0, stream>>>(
        z, nullptr, br, bc, W1b, b1, W2b, b2, out, n_edges);
  }
}
